// CRF_72851235275262
// MI455X (gfx1250) — matrix operand order repaired and checked by reading, not compiled, not run on hardware
//
#include <hip/hip_runtime.h>
#include <hip/hip_bf16.h>
#include <math.h>

// CRF forward-algorithm loss on gfx1250.
// logsumexp_j(alpha_j + trans_jk) = m + log( sum_j exp(alpha_j-m) * E_jk ),
// E = exp(trans) precomputed -> each recursion step is a (16x64)@(64x64) matmul
// per wave, done as S^T = E^T (A operand) x p^T (B operand) with
// v_wmma_f32_16x16x32_f16. Cross-lane traffic uses v_permlanex16_b32 (VALU),
// transcendentals use raw v_exp_f32/v_log_f32 (inputs are range-safe),
// masking is a branchless bit-select (no exec-mask control flow in the loop).

typedef __attribute__((ext_vector_type(16))) _Float16     v16h;
typedef __attribute__((ext_vector_type(8)))  float        v8f;
typedef __attribute__((ext_vector_type(8)))  unsigned int ui8;

#define Tn 512
#define Bn 1024
#define Nn 64

#define LOG2E 1.44269504088896340736f
#define LN2   0.69314718055994530942f

__device__ __forceinline__ float fast_exp(float x) {
    return __builtin_amdgcn_exp2f(x * LOG2E);          // v_exp_f32, no slow path
}
__device__ __forceinline__ float fast_log(float x) {
    return __builtin_amdgcn_logf(x) * LN2;             // v_log_f32, no slow path
}
__device__ __forceinline__ unsigned pack_h2(float a, float b) {
    return __builtin_bit_cast(unsigned,
        __builtin_amdgcn_cvt_pkrtz(a, b));             // v_cvt_pkrtz_f16_f32
}
// branchless exact select via bit ops (lowers to v_bfi / bitop3)
__device__ __forceinline__ float bitsel_f(unsigned sel, float a, float b) {
    unsigned ua = __builtin_bit_cast(unsigned, a);
    unsigned ub = __builtin_bit_cast(unsigned, b);
    return __builtin_bit_cast(float, (ua & sel) | (ub & ~sel));
}
// lane <-> lane^16 exchange (row swap) as pure VALU: v_permlanex16_b32
__device__ __forceinline__ unsigned swap16u(unsigned x) {
    return (unsigned)__builtin_amdgcn_permlanex16(
        (int)x, (int)x, 0x76543210, 0xfedcba98, false, false);
}
__device__ __forceinline__ float swap16f(float x) {
    int i = __builtin_bit_cast(int, x);
    i = __builtin_amdgcn_permlanex16(i, i, 0x76543210, 0xfedcba98, false, false);
    return __builtin_bit_cast(float, i);
}
// xor-k permutation within each 16-lane row: v_permlane16_b32
__device__ __forceinline__ float permrow_f(float x, int s1, int s2) {
    int i = __builtin_bit_cast(int, x);
    i = __builtin_amdgcn_permlane16(i, i, s1, s2, false, false);
    return __builtin_bit_cast(float, i);
}

// ---------------------------------------------------------------------------
// Main kernel: one wave32 owns 16 batch rows, scans all T steps.
// alpha is kept transposed in WMMA C/D layout, 4 tiles of (k 16) x (m 16):
//   tile c, VGPR r, lane l  <->  k = 16c + r + 8*(l>>4),  m = l & 15.
// Per-lane: row max/mask/emit are uniform in m (m = lane&15), k chunks contiguous.
// ---------------------------------------------------------------------------
__global__ __launch_bounds__(128) void crf_forward_kernel(
    const float* __restrict__ emit, const float* __restrict__ trans,
    const float* __restrict__ strans, const float* __restrict__ etrans,
    const unsigned char* __restrict__ mask, float* __restrict__ partials)
{
    __shared__ float ldsT[Nn * Nn];

    const int tid = threadIdx.x;
    for (int i = tid; i < Nn * Nn; i += 128) ldsT[i] = trans[i];
    __syncthreads();

    const int wave = tid >> 5;
    const int lane = tid & 31;
    const int m    = lane & 15;   // batch sub-row owned by this lane
    const int h    = lane >> 4;   // half-wave
    const int b    = (blockIdx.x * 4 + wave) * 16 + m;

    // ---- Preload E^T = exp(trans)^T as 8 A-operand fragments (f16). ----
    // A frag for k-tile c, j-chunk J (j in [32J, 32J+32)):
    //   lane row M = m -> k = 16c + m;  halfs [0..7] = j = 32J+8h+i, [8..15] = j = 32J+16+8h+i.
    v16h Efrag[4][2];
    #pragma unroll
    for (int c = 0; c < 4; ++c) {
        const int krow = 16 * c + m;
        #pragma unroll
        for (int J = 0; J < 2; ++J) {
            v16h a;
            const int jb = 32 * J + 8 * h;
            #pragma unroll
            for (int i = 0; i < 8; ++i) {
                a[i]     = (_Float16)fast_exp(ldsT[(jb + i) * Nn + krow]);
                a[i + 8] = (_Float16)fast_exp(ldsT[(jb + 16 + i) * Nn + krow]);
            }
            Efrag[c][J] = a;
        }
    }

    // ---- Init alpha: alpha(m,k) = strans[k] + emit[0,b,k]. ----
    v8f alphaT[4];
    {
        const float* erow0 = emit + (size_t)b * Nn;   // t = 0
        #pragma unroll
        for (int c = 0; c < 4; ++c) {
            const int kb = 16 * c + 8 * h;
            float4 s0 = *(const float4*)(strans + kb);
            float4 s1 = *(const float4*)(strans + kb + 4);
            float4 e0 = *(const float4*)(erow0 + kb);
            float4 e1 = *(const float4*)(erow0 + kb + 4);
            alphaT[c][0] = s0.x + e0.x; alphaT[c][1] = s0.y + e0.y;
            alphaT[c][2] = s0.z + e0.z; alphaT[c][3] = s0.w + e0.w;
            alphaT[c][4] = s1.x + e1.x; alphaT[c][5] = s1.y + e1.y;
            alphaT[c][6] = s1.z + e1.z; alphaT[c][7] = s1.w + e1.w;
        }
    }

    // ---- Serial scan over time. ----
    for (int t = 1; t < Tn; ++t) {
        const float* erow = emit + ((size_t)t * Bn + b) * Nn;

        // Issue all memory for this step up front (emit row chunks + mask byte).
        float4 ee[4][2];
        #pragma unroll
        for (int c = 0; c < 4; ++c) {
            const int kb = 16 * c + 8 * h;
            ee[c][0] = *(const float4*)(erow + kb);
            ee[c][1] = *(const float4*)(erow + kb + 4);
        }
        const unsigned sel = (mask[(size_t)t * Bn + b] != 0) ? 0xFFFFFFFFu : 0u;
        if (t + 1 < Tn)
            __builtin_prefetch(emit + ((size_t)(t + 1) * Bn + b) * Nn, 0, 0);

        // per-batch-row max over k (per-lane 32 values, then swap halves)
        float mx = alphaT[0][0];
        #pragma unroll
        for (int c = 0; c < 4; ++c)
            #pragma unroll
            for (int r = 0; r < 8; ++r) mx = fmaxf(mx, alphaT[c][r]);
        mx = fmaxf(mx, swap16f(mx));

        // p = exp(alpha - mx), packed f16x2 per tile (dword d = k pair 2d,2d+1)
        unsigned pc[4][4];
        #pragma unroll
        for (int c = 0; c < 4; ++c)
            #pragma unroll
            for (int d = 0; d < 4; ++d)
                pc[c][d] = pack_h2(fast_exp(alphaT[c][2 * d] - mx),
                                   fast_exp(alphaT[c][2 * d + 1] - mx));

        // Build B fragments p^T (64x16): frag J, lane l: col m, halfs [0..7] j = 32J+8h+i, [8..15] j = 32J+16+8h+i.
        //   halfs 0..7 own tile c0, 8..15 own tile c1
        v16h Bfrag[2];
        #pragma unroll
        for (int J = 0; J < 2; ++J) {
            const int c0 = 2 * J, c1 = 2 * J + 1;
            ui8 u;
            #pragma unroll
            for (int d = 0; d < 4; ++d) {
                unsigned own0 = pc[c0][d], own1 = pc[c1][d];
                unsigned r0 = swap16u(own0);
                unsigned r1 = swap16u(own1);
                u[d]     = own0;
                u[d + 4] = own1;
            }
            Bfrag[J] = __builtin_bit_cast(v16h, u);
        }

        // All 8 WMMAs back-to-back: S^T tiles, K = 64 as two K=32 WMMAs each.
        v8f accs[4];
        #pragma unroll
        for (int c = 0; c < 4; ++c) {
            v8f acc = {};
            acc = __builtin_amdgcn_wmma_f32_16x16x32_f16(
                false, Efrag[c][0], false, Bfrag[0], (short)0, acc, false, false);
            accs[c] = __builtin_amdgcn_wmma_f32_16x16x32_f16(
                false, Efrag[c][1], false, Bfrag[1], (short)0, acc, false, false);
        }

        // Branchless update: alpha = sel ? (mx + log(S) + emit) : alpha
        #pragma unroll
        for (int c = 0; c < 4; ++c) {
            float ev[8] = {ee[c][0].x, ee[c][0].y, ee[c][0].z, ee[c][0].w,
                           ee[c][1].x, ee[c][1].y, ee[c][1].z, ee[c][1].w};
            #pragma unroll
            for (int r = 0; r < 8; ++r) {
                float s = mx + fast_log(accs[c][r]) + ev[r];
                alphaT[c][r] = bitsel_f(sel, s, alphaT[c][r]);
            }
        }
    }

    // ---- logsumexp_k(alpha + etrans) per batch row, summed over this wave's rows ----
    float vv[4][8];
    float fmx = -INFINITY;
    #pragma unroll
    for (int c = 0; c < 4; ++c) {
        const int kb = 16 * c + 8 * h;
        float4 t0 = *(const float4*)(etrans + kb);
        float4 t1 = *(const float4*)(etrans + kb + 4);
        float tv[8] = {t0.x, t0.y, t0.z, t0.w, t1.x, t1.y, t1.z, t1.w};
        #pragma unroll
        for (int r = 0; r < 8; ++r) {
            vv[c][r] = alphaT[c][r] + tv[r];
            fmx = fmaxf(fmx, vv[c][r]);
        }
    }
    fmx = fmaxf(fmx, swap16f(fmx));
    float ssum = 0.f;
    #pragma unroll
    for (int c = 0; c < 4; ++c)
        #pragma unroll
        for (int r = 0; r < 8; ++r) ssum += fast_exp(vv[c][r] - fmx);
    ssum += swap16f(ssum);
    float lse = fmx + fast_log(ssum);      // duplicated in lanes m and m+16
    float contrib = h ? 0.f : lse;         // count each row once
    // xor-reduce within the 16-lane row via v_permlane16_b32
    contrib += permrow_f(contrib, 0x67452301, 0xefcdab89);  // xor 1
    contrib += permrow_f(contrib, 0x45670123, 0xcdef89ab);  // xor 2
    contrib += permrow_f(contrib, 0x01234567, 0x89abcdef);  // xor 4
    contrib += permrow_f(contrib, 0xfedcba98, 0x76543210);  // xor 8
    if (lane == 0) partials[blockIdx.x * 4 + wave] = contrib;
}

// ---------------------------------------------------------------------------
// Gold path score: one thread per batch element.
// ---------------------------------------------------------------------------
__global__ __launch_bounds__(256) void crf_score_kernel(
    const float* __restrict__ emit, const float* __restrict__ trans,
    const float* __restrict__ strans, const float* __restrict__ etrans,
    const int* __restrict__ target, const unsigned char* __restrict__ mask,
    float* __restrict__ partials)
{
    __shared__ float red[256];
    const int b = blockIdx.x * 256 + threadIdx.x;

    int prev = target[b];                               // t = 0 always valid (len >= 1)
    float s  = emit[(size_t)b * Nn + prev] + strans[prev];
    int last = prev;
    for (int t = 1; t < Tn; ++t) {
        if (mask[(size_t)t * Bn + b]) {
            int cur = target[(size_t)t * Bn + b];
            s += emit[((size_t)t * Bn + b) * Nn + cur] + trans[prev * Nn + cur];
            prev = cur; last = cur;
        }
    }
    s += etrans[last];

    red[threadIdx.x] = s;
    __syncthreads();
    for (int off = 128; off > 0; off >>= 1) {
        if (threadIdx.x < off) red[threadIdx.x] += red[threadIdx.x + off];
        __syncthreads();
    }
    if (threadIdx.x == 0) partials[64 + blockIdx.x] = red[0];
}

// ---------------------------------------------------------------------------
// Final deterministic reduce: out = (logZ - score) / B
// ---------------------------------------------------------------------------
__global__ __launch_bounds__(64) void crf_final_kernel(
    const float* __restrict__ partials, float* __restrict__ out)
{
    __shared__ float red[64];
    const int t = threadIdx.x;
    red[t] = partials[t];                  // 64 logZ wave partials
    __syncthreads();
    for (int off = 32; off > 0; off >>= 1) {
        if (t < off) red[t] += red[t + off];
        __syncthreads();
    }
    if (t == 0) {
        float score = partials[64] + partials[65] + partials[66] + partials[67];
        out[0] = (red[0] - score) / (float)Bn;
    }
}

extern "C" void kernel_launch(void* const* d_in, const int* in_sizes, int n_in,
                              void* d_out, int out_size, void* d_ws, size_t ws_size,
                              hipStream_t stream) {
    (void)in_sizes; (void)n_in; (void)out_size; (void)ws_size;
    const float*         emit   = (const float*)d_in[0];
    const float*         trans  = (const float*)d_in[1];
    const float*         strans = (const float*)d_in[2];
    const float*         etrans = (const float*)d_in[3];
    const int*           target = (const int*)d_in[4];
    const unsigned char* mask   = (const unsigned char*)d_in[5];
    float* ws  = (float*)d_ws;
    float* out = (float*)d_out;

    crf_forward_kernel<<<16, 128, 0, stream>>>(emit, trans, strans, etrans, mask, ws);
    crf_score_kernel<<<4, 256, 0, stream>>>(emit, trans, strans, etrans, target, mask, ws);
    crf_final_kernel<<<1, 64, 0, stream>>>(ws, out);
}
